// interact_22874995818903
// MI455X (gfx1250) — matrix operand order repaired and checked by reading, not compiled, not run on hardware
//
#include <hip/hip_runtime.h>

#define DEVINL __device__ __forceinline__

typedef __attribute__((ext_vector_type(16))) _Float16 v16h;
typedef __attribute__((ext_vector_type(8)))  float    v8f;

constexpr int Bc = 8, Sc = 128, Hc = 768, NSPK = 8;
constexpr int H2 = 2 * Hc, H3 = 3 * Hc, H6 = 6 * Hc;
constexpr int TOK = Bc * Sc;              // 1024 tokens
constexpr int NBLK = 32, NTHR = 256;      // persistent kernels: 32 WGs x 8 waves
constexpr float RSCALE = 0.03608439182435161f;  // 1/sqrt(768)

// ---------------- CDNA5 async global->LDS path (inline asm) ----------------
// 16-byte async copy global -> LDS, tracked by ASYNCcnt (ISA 15.18.3 opcode 98).
// GV mode: vdst = LDS byte address (low 32 bits of the generic shared pointer,
// per the ISA 10.2 LDS-aperture truncation rule), vaddr = 64-bit global address.
static DEVINL void async_ld16(void* lds, const void* g)
{
  unsigned ldsa = (unsigned)(unsigned long long)lds;
  unsigned long long ga = (unsigned long long)g;
  asm volatile("global_load_async_to_lds_b128 %0, %1, off"
               :: "v"(ldsa), "v"(ga)
               : "memory");
}

static DEVINL void async_wait0()
{
  asm volatile("s_wait_asynccnt 0x0" ::: "memory");
}

// ---------------- workspace layout ----------------
// f16 region (units: _Float16)
constexpr size_t OH_BERT  = 0;
constexpr size_t OH_DIAH  = OH_BERT  + (size_t)TOK * Hc;
constexpr size_t OH_HH    = OH_DIAH  + (size_t)TOK * Hc;      // rnnD state, 16xH (rows 8..15 zero)
constexpr size_t OH_DWIH  = OH_HH    + (size_t)16 * Hc;
constexpr size_t OH_DWHH  = OH_DWIH  + (size_t)H3 * Hc;
constexpr size_t OH_SWIH  = OH_DWHH  + (size_t)H3 * Hc;
constexpr size_t OH_SWHH  = OH_SWIH  + (size_t)H6 * H2;
constexpr size_t OH_GWIH  = OH_SWHH  + (size_t)H6 * H2;
constexpr size_t OH_GWHH  = OH_GWIH  + (size_t)H3 * H3;
constexpr size_t OH_GCOMB = OH_GWHH  + (size_t)H3 * Hc;
constexpr size_t OH_WQ    = OH_GCOMB + (size_t)H3 * Hc;
constexpr size_t OH_WK    = OH_WQ    + (size_t)Hc * Hc;
constexpr size_t OH_WV    = OH_WK    + (size_t)Hc * Hc;
constexpr size_t OH_END   = OH_WV    + (size_t)Hc * Hc;
constexpr size_t FBASE_BYTES = ((OH_END * 2 + 255) / 256) * 256;
// f32 region (units: float), base = ws + FBASE_BYTES
constexpr size_t OF_GID = 0;
constexpr size_t OF_QU  = OF_GID + (size_t)TOK * H3;
constexpr size_t OF_SIU = OF_QU  + (size_t)TOK * Hc;
constexpr size_t OF_GIU = OF_SIU + (size_t)TOK * H6;
constexpr size_t OF_DIA = OF_GIU + (size_t)TOK * H3;
constexpr size_t OF_HF  = OF_DIA + (size_t)TOK * Hc;     // rnnD f32 state, BxH
constexpr size_t OF_GHW = OF_HF  + (size_t)Bc * Hc;      // 16 x 3H gh scratch
constexpr size_t OF_KC  = OF_GHW + (size_t)16 * H3;      // K cache SxH
constexpr size_t OF_VC  = OF_KC  + (size_t)Sc * Hc;      // V cache SxH
constexpr size_t OF_SC  = OF_VC  + (size_t)Sc * Hc;      // scores (128)
constexpr size_t OF_AP  = OF_SC  + 128;                  // attn_part (H)
constexpr size_t OF_GIS = OF_AP  + Hc;                   // gi_s (6H)
constexpr size_t OF_GHS = OF_GIS + H6;                   // gh_s (6H)
constexpr size_t OF_GIG = OF_GHS + H6;                   // gi_g (3H)
constexpr size_t OF_GHG = OF_GIG + H3;                   // gh_g (3H)
constexpr size_t OF_UG  = OF_GHG + H3;                   // u_g (2H)
constexpr size_t OF_DH  = OF_UG  + H2;                   // dh / op_s (H)
constexpr size_t OF_SP  = OF_DH  + Hc;                   // speaker states NSPK x 2H
constexpr size_t OF_END = OF_SP  + (size_t)NSPK * H2;
constexpr size_t FREG_BYTES = ((OF_END * 4 + 255) / 256) * 256;
constexpr size_t BAR_OFF = FBASE_BYTES + FREG_BYTES;

// ---------------- small helpers ----------------
static DEVINL float sigm(float x) { return 1.f / (1.f + expf(-x)); }

// grid-wide sense-reversing barrier (persistent cooperative kernels)
static DEVINL void gbar(unsigned* bar, unsigned nb)
{
  __syncthreads();
  if (threadIdx.x == 0) {
    __threadfence();
    unsigned g0 = atomicAdd(&bar[1], 0u);
    unsigned a  = atomicAdd(&bar[0], 1u);
    if (a == nb - 1u) {
      atomicExch(&bar[0], 0u);
      __threadfence();
      atomicAdd(&bar[1], 1u);
    } else {
      while (atomicAdd(&bar[1], 0u) == g0) __builtin_amdgcn_s_sleep(8);
    }
    __threadfence();
  }
  __syncthreads();
}

// y[n0..n0+15] = W[n0..n0+15, 0..K) * x + addv[n0..n0+15]
// One wave; W f16 row-major (ldw), x f32 vector; uses v_wmma with B column 0.
static DEVINL void matvec16(const _Float16* __restrict__ W, int ldw, int n0,
                            const float* __restrict__ x, int K,
                            const float* __restrict__ addv,
                            float* __restrict__ y, int lane)
{
  const int mrow = lane & 15;
  const int kbA  = (lane >> 4) * 16;
  const int kbB  = (lane >> 4) * 16;
  const _Float16* wrow = W + (size_t)(n0 + mrow) * ldw;
  v8f acc = {};
  for (int k0 = 0; k0 < K; k0 += 32) {
    v16h af, bf;
#pragma unroll
    for (int i = 0; i < 8; ++i) {
      af[i]     = wrow[k0 + kbA + i];
      af[i + 8] = wrow[k0 + kbA + 8 + i];
    }
#pragma unroll
    for (int i = 0; i < 16; ++i) bf[i] = (_Float16)0.f;
    if (mrow == 0) {
      const float* xp = x + k0 + kbB;
#pragma unroll
      for (int i = 0; i < 16; ++i) bf[i] = (_Float16)xp[i];
    }
    __builtin_prefetch(wrow + k0 + 32, 0, 1);
    acc = __builtin_amdgcn_wmma_f32_16x16x32_f16(false, af, false, bf,
                                                 (short)0, acc, false, false);
  }
  if (mrow == 0) {
    const int rb = (lane >> 4) * 8;
#pragma unroll
    for (int j = 0; j < 8; ++j) y[n0 + rb + j] = acc[j] + addv[n0 + rb + j];
  }
}

// ---------------- prep kernels ----------------
__global__ void __launch_bounds__(NTHR) k_zero_init(_Float16* hh, float* hf, unsigned* bar)
{
  int g = blockIdx.x * blockDim.x + threadIdx.x;
  int gs = gridDim.x * blockDim.x;
  for (int i = g; i < 16 * Hc; i += gs) hh[i] = (_Float16)0.f;
  for (int i = g; i < Bc * Hc; i += gs) hf[i] = 0.f;
  if (g < 2) bar[g] = 0u;
}

__global__ void __launch_bounds__(NTHR) k_cvt(_Float16* __restrict__ dst,
                                              const float* __restrict__ src, int n)
{
  int g = blockIdx.x * blockDim.x + threadIdx.x;
  int gs = gridDim.x * blockDim.x;
  for (int i = g; i < n; i += gs) dst[i] = (_Float16)src[i];
}

// gcomb[n,k] = g_wih[n, H+k] + g_wih[n, 2H+k]   (3H x H), f16
__global__ void __launch_bounds__(NTHR) k_gcomb(_Float16* __restrict__ dst,
                                                const float* __restrict__ g_wih)
{
  int g = blockIdx.x * blockDim.x + threadIdx.x;
  int gs = gridDim.x * blockDim.x;
  for (int i = g; i < H3 * Hc; i += gs) {
    int r = i / Hc, k = i - r * Hc;
    dst[i] = (_Float16)(g_wih[(size_t)r * H3 + Hc + k] + g_wih[(size_t)r * H3 + 2 * Hc + k]);
  }
}

__global__ void __launch_bounds__(NTHR) k_gather(_Float16* __restrict__ berth,
                                                 const float* __restrict__ emb,
                                                 const int* __restrict__ inputs)
{
  int g = blockIdx.x * blockDim.x + threadIdx.x;
  int gs = gridDim.x * blockDim.x;
  for (int i = g; i < TOK * Hc; i += gs) {
    int t = i / Hc, k = i - t * Hc;
    berth[i] = (_Float16)emb[(size_t)inputs[t] * Hc + k];
  }
}

// ---------------- token-parallel WMMA GEMM: C = A * W^T + bias ----------------
// A: [M,K] f16 row-major (M = gridDim.y*16); W: [N,K] f16 (row stride ldw); C f32.
// Block = 8 waves covering a 16(M) x 128(N) macro tile. K is staged through LDS in
// 32-wide slices with double-buffered async global->LDS copies (ASYNCcnt).
__global__ void __launch_bounds__(NTHR) k_gemm(const _Float16* __restrict__ A, int lda,
                                               const _Float16* __restrict__ W, int ldw,
                                               const float* __restrict__ bias,
                                               float* __restrict__ C, int ldc,
                                               int N, int K)
{
  __shared__ __align__(16) _Float16 As[2][16 * 32];
  __shared__ __align__(16) _Float16 Ws[2][128 * 32];
  const int tid = threadIdx.x;
  const int wave = tid >> 5, lane = tid & 31;
  const int m0 = blockIdx.y * 16;
  const int nB = blockIdx.x * 128;
  const int mrow = lane & 15, kbA = (lane >> 4) * 16, kbB = (lane >> 4) * 16;

  // 64 A-chunks + 512 W-chunks of 16B per K-slice, spread over 256 threads
  auto issue = [&](int k0, int buf) {
    for (int id = tid; id < 576; id += NTHR) {
      if (id < 64) {
        int r = id >> 2, c = (id & 3) * 8;
        async_ld16(&As[buf][r * 32 + c], A + (size_t)(m0 + r) * lda + k0 + c);
      } else {
        int wid = id - 64;
        int r = wid >> 2, c = (wid & 3) * 8;
        async_ld16(&Ws[buf][r * 32 + c], W + (size_t)(nB + r) * ldw + k0 + c);
      }
    }
  };

  v8f acc = {};
  const int nk = K / 32;
  issue(0, 0);
  for (int kc = 0; kc < nk; ++kc) {
    async_wait0();
    __syncthreads();                         // slice kc resident in LDS for all waves
    if (kc + 1 < nk) issue((kc + 1) * 32, (kc + 1) & 1);  // overlap next slice
    const _Float16* as = &As[kc & 1][0];
    const _Float16* ws = &Ws[kc & 1][wave * 16 * 32];
    v16h af, bf;
#pragma unroll
    for (int i = 0; i < 8; ++i) {
      af[i]     = as[mrow * 32 + kbA + i];
      af[i + 8] = as[mrow * 32 + kbA + 8 + i];
    }
#pragma unroll
    for (int i = 0; i < 16; ++i) bf[i] = ws[mrow * 32 + kbB + i];
    acc = __builtin_amdgcn_wmma_f32_16x16x32_f16(false, af, false, bf,
                                                 (short)0, acc, false, false);
  }
  const int n0 = nB + wave * 16;
  if (n0 < N) {
    const int cn = n0 + mrow;
    const int rb = m0 + (lane >> 4) * 8;
    const float bsv = bias[cn];
#pragma unroll
    for (int j = 0; j < 8; ++j) C[(size_t)(rb + j) * ldc + cn] = acc[j] + bsv;
  }
}

// ---------------- persistent rnnD scan (128 steps) ----------------
__global__ void __launch_bounds__(NTHR) k_rnnd(_Float16* __restrict__ hh,
                                               const _Float16* __restrict__ dwhh,
                                               const float* __restrict__ GId,
                                               const float* __restrict__ d_bhh,
                                               float* __restrict__ hf,
                                               float* __restrict__ ghw,
                                               float* __restrict__ dia,
                                               _Float16* __restrict__ diah,
                                               unsigned* bar)
{
  const int tid = threadIdx.x, bid = blockIdx.x;
  const int gtid = bid * NTHR + tid, gsz = gridDim.x * NTHR;
  const int lane = tid & 31;
  const int gw = gtid >> 5, nw = gsz >> 5;
  for (int s = 0; s < Sc; ++s) {
    // gh = h @ d_whh^T, 16x3H (rows 8..15 of h are zero)
    for (int nt = gw; nt < H3 / 16; nt += nw) {
      const int n0 = nt * 16;
      const int mrow = lane & 15, kbA = (lane >> 4) * 16, kbB = (lane >> 4) * 16;
      const _Float16* arow = hh + (size_t)mrow * Hc;
      const _Float16* wrow = dwhh + (size_t)(n0 + mrow) * Hc;
      v8f acc = {};
      for (int k0 = 0; k0 < Hc; k0 += 32) {
        v16h af, bf;
#pragma unroll
        for (int i = 0; i < 8; ++i) {
          af[i]     = arow[k0 + kbA + i];
          af[i + 8] = arow[k0 + kbA + 8 + i];
        }
#pragma unroll
        for (int i = 0; i < 16; ++i) bf[i] = wrow[k0 + kbB + i];
        __builtin_prefetch(wrow + k0 + 32, 0, 1);
        acc = __builtin_amdgcn_wmma_f32_16x16x32_f16(false, af, false, bf,
                                                     (short)0, acc, false, false);
      }
      const int cn = n0 + mrow, rb = (lane >> 4) * 8;
#pragma unroll
      for (int j = 0; j < 8; ++j) ghw[(size_t)(rb + j) * H3 + cn] = acc[j];
    }
    gbar(bar, gridDim.x);
    // pointwise GRU
    for (int i = gtid; i < Bc * Hc; i += gsz) {
      const int bb = i / Hc, j = i - bb * Hc;
      const float* gi = GId + ((size_t)bb * Sc + s) * H3;
      const float* gh = ghw + (size_t)bb * H3;
      float r = sigm(gi[j] + gh[j] + d_bhh[j]);
      float z = sigm(gi[Hc + j] + gh[Hc + j] + d_bhh[Hc + j]);
      float nn = tanhf(gi[2 * Hc + j] + r * (gh[2 * Hc + j] + d_bhh[2 * Hc + j]));
      float hn = (1.f - z) * nn + z * hf[i];
      hf[i] = hn;
      hh[i] = (_Float16)hn;
      size_t o = ((size_t)bb * Sc + s) * Hc + j;
      dia[o] = hn;
      diah[o] = (_Float16)hn;
    }
    gbar(bar, gridDim.x);
  }
}

// ---------------- persistent main loop (1024 steps) ----------------
__global__ void __launch_bounds__(NTHR) k_mainloop(
    const _Float16* __restrict__ swih, const _Float16* __restrict__ swhh,
    const _Float16* __restrict__ gwih, const _Float16* __restrict__ gwhh,
    const _Float16* __restrict__ wkh,  const _Float16* __restrict__ wvh,
    const float* __restrict__ QU,  const float* __restrict__ SIU,
    const float* __restrict__ GIU, const float* __restrict__ dia,
    float* __restrict__ Kc, float* __restrict__ Vc,
    float* __restrict__ scores, float* __restrict__ attnp,
    float* __restrict__ gis, float* __restrict__ ghs,
    float* __restrict__ gig, float* __restrict__ ghg,
    float* __restrict__ ug, float* __restrict__ dh,
    float* __restrict__ sp,
    const int* __restrict__ chat_ids, const int* __restrict__ spk_info,
    const float* __restrict__ s_bhh, const float* __restrict__ g_bhh,
    const float* __restrict__ bk, const float* __restrict__ bv,
    const float* __restrict__ h0, const float* __restrict__ d_h0,
    const float* __restrict__ attn_h,
    float* __restrict__ out_op, float* __restrict__ out_spop,
    unsigned* bar)
{
  const int tid = threadIdx.x, bid = blockIdx.x;
  const int gtid = bid * NTHR + tid, gsz = gridDim.x * NTHR;
  const int lane = tid & 31;
  const int gw = gtid >> 5, nw = gsz >> 5;
  __shared__ float sprob[Sc];

  for (int b = 0; b < Bc; ++b) {
    const int cid = chat_ids[b];
    // per-batch init: K/V rows = projection of zero rows (= biases), speaker states = h0
    for (int i = gtid; i < Sc * Hc; i += gsz) {
      int k = i % Hc;
      Kc[i] = bk[k]; Vc[i] = bv[k];
    }
    for (int i = gtid; i < NSPK * H2; i += gsz) sp[i] = h0[i % H2];
    if (b == 0) for (int i = gtid; i < Hc; i += gsz) dh[i] = d_h0[i];
    gbar(bar, gridDim.x);

    for (int s = 0; s < Sc; ++s) {
      const size_t tok = (size_t)b * Sc + s;
      const float* q = QU + tok * Hc;      // precomputed query projection
      // stage A: scores_j = (K_j . q)/sqrt(H), j <= s (row s holds bk = zero-row proj)
      for (int j = gw; j <= s; j += nw) {
        const float* kr = Kc + (size_t)j * Hc;
        float p = 0.f;
        for (int k = lane; k < Hc; k += 32) p += kr[k] * q[k];
        for (int off = 16; off > 0; off >>= 1) p += __shfl_xor(p, off, 32);
        if (lane == 0) scores[j] = p * RSCALE;
      }
      gbar(bar, gridDim.x);
      // stage B: redundant per-block softmax, then each block's slice of attn columns
      {
        float mx = -3.4e38f;
        for (int j = 0; j <= s; ++j) mx = fmaxf(mx, scores[j]);
        float sum = 0.f;
        for (int j = 0; j <= s; ++j) sum += expf(scores[j] - mx);
        float inv = 1.f / sum;
        for (int j = tid; j <= s; j += NTHR) sprob[j] = expf(scores[j] - mx) * inv;
        __syncthreads();
        const int cw = Hc / NBLK;               // 24 columns per block
        const int col = bid * cw + tid;
        if (tid < cw) {
          float a = 0.f;
          for (int j = 0; j <= s; ++j) a += sprob[j] * Vc[(size_t)j * Hc + col];
          attnp[col] = (s == 0) ? attn_h[col] : a;
        }
      }
      gbar(bar, gridDim.x);
      // stage 1: gi_s = s_wih[:, :H]@attn + SIU ; gh_s = s_whh@h_spk + s_bhh
      const int spk = spk_info[cid * Sc + s];
      const float* hspk = sp + (size_t)spk * H2;
      for (int t = gw; t < 2 * (H6 / 16); t += nw) {
        if (t < H6 / 16)
          matvec16(swih, H2, t * 16, attnp, Hc, SIU + tok * H6, gis, lane);
        else
          matvec16(swhh, H2, (t - H6 / 16) * 16, hspk, H2, s_bhh, ghs, lane);
      }
      gbar(bar, gridDim.x);
      // stage 2: GRU-S pointwise; spop out; u_g = [spop[:H], h_new[H:]]
      const float* utt = dia + tok * Hc;
      float* spw = sp + (size_t)spk * H2;
      for (int j = gtid; j < H2; j += gsz) {
        float r = sigm(gis[j] + ghs[j]);
        float z = sigm(gis[H2 + j] + ghs[H2 + j]);
        float nn = tanhf(gis[2 * H2 + j] + r * ghs[2 * H2 + j]);
        float hn = (1.f - z) * nn + z * hspk[j];
        float tj = (j < Hc) ? attnp[j] : utt[j - Hc];
        float so = hn + tj;
        out_spop[tok * H2 + j] = so;
        spw[j] = hn;
        ug[j] = (j < Hc) ? so : hn;
      }
      gbar(bar, gridDim.x);
      // stage 3: gi_g = g_wih[:, :2H]@u_g + GIU ; gh_g = g_whh@dh + g_bhh
      for (int t = gw; t < 2 * (H3 / 16); t += nw) {
        if (t < H3 / 16)
          matvec16(gwih, H3, t * 16, ug, H2, GIU + tok * H3, gig, lane);
        else
          matvec16(gwhh, Hc, (t - H3 / 16) * 16, dh, Hc, g_bhh, ghg, lane);
      }
      gbar(bar, gridDim.x);
      // stage 4: GRU-G pointwise; op out; dh := op
      for (int j = gtid; j < Hc; j += gsz) {
        float r = sigm(gig[j] + ghg[j]);
        float z = sigm(gig[Hc + j] + ghg[Hc + j]);
        float nn = tanhf(gig[2 * Hc + j] + r * ghg[2 * Hc + j]);
        float o = (1.f - z) * nn + z * dh[j];
        out_op[tok * Hc + j] = o;
        dh[j] = o;
      }
      gbar(bar, gridDim.x);
      // stage 5: incremental K/V rows for the new op
      for (int t = gw; t < 2 * (Hc / 16); t += nw) {
        if (t < Hc / 16)
          matvec16(wkh, Hc, t * 16, dh, Hc, bk, Kc + (size_t)s * Hc, lane);
        else
          matvec16(wvh, Hc, (t - Hc / 16) * 16, dh, Hc, bv, Vc + (size_t)s * Hc, lane);
      }
      gbar(bar, gridDim.x);
    }
  }
}

// ---------------- host launcher ----------------
extern "C" void kernel_launch(void* const* d_in, const int* in_sizes, int n_in,
                              void* d_out, int out_size, void* d_ws, size_t ws_size,
                              hipStream_t stream)
{
  (void)in_sizes; (void)n_in; (void)out_size; (void)ws_size;
  const int*   chat_ids = (const int*)d_in[0];
  const int*   spk_info = (const int*)d_in[1];
  const int*   inputs   = (const int*)d_in[4];
  const float* emb   = (const float*)d_in[5];
  const float* d_wih = (const float*)d_in[6];
  const float* d_whh = (const float*)d_in[7];
  const float* d_bih = (const float*)d_in[8];
  const float* d_bhh = (const float*)d_in[9];
  const float* g_wih = (const float*)d_in[10];
  const float* g_whh = (const float*)d_in[11];
  const float* g_bih = (const float*)d_in[12];
  const float* g_bhh = (const float*)d_in[13];
  const float* s_wih = (const float*)d_in[14];
  const float* s_whh = (const float*)d_in[15];
  const float* s_bih = (const float*)d_in[16];
  const float* s_bhh = (const float*)d_in[17];
  const float* wq = (const float*)d_in[18];
  const float* bq = (const float*)d_in[19];
  const float* wk = (const float*)d_in[20];
  const float* bk = (const float*)d_in[21];
  const float* wv = (const float*)d_in[22];
  const float* bv = (const float*)d_in[23];
  const float* h0 = (const float*)d_in[24];
  const float* d_h0 = (const float*)d_in[25];
  const float* attn_h = (const float*)d_in[26];

  char* wsb = (char*)d_ws;
  _Float16* Hb = (_Float16*)wsb;
  float* Fb = (float*)(wsb + FBASE_BYTES);
  unsigned* bar = (unsigned*)(wsb + BAR_OFF);

  _Float16 *berth = Hb + OH_BERT, *diah = Hb + OH_DIAH, *hh = Hb + OH_HH,
           *dwihh = Hb + OH_DWIH, *dwhhh = Hb + OH_DWHH,
           *swihh = Hb + OH_SWIH, *swhhh = Hb + OH_SWHH,
           *gwihh = Hb + OH_GWIH, *gwhhh = Hb + OH_GWHH, *gcomb = Hb + OH_GCOMB,
           *wqh = Hb + OH_WQ, *wkh = Hb + OH_WK, *wvh = Hb + OH_WV;
  float *GId = Fb + OF_GID, *QU = Fb + OF_QU, *SIU = Fb + OF_SIU, *GIU = Fb + OF_GIU,
        *dia = Fb + OF_DIA, *hf = Fb + OF_HF, *ghw = Fb + OF_GHW,
        *Kc = Fb + OF_KC, *Vc = Fb + OF_VC, *scores = Fb + OF_SC, *attnp = Fb + OF_AP,
        *gis = Fb + OF_GIS, *ghs = Fb + OF_GHS, *gig = Fb + OF_GIG, *ghg = Fb + OF_GHG,
        *ug = Fb + OF_UG, *dh = Fb + OF_DH, *sp = Fb + OF_SP;

  float* out_op = (float*)d_out;
  float* out_spop = out_op + (size_t)TOK * Hc;

  k_zero_init<<<64, NTHR, 0, stream>>>(hh, hf, bar);
  // f32 -> f16 weight conversions
  k_cvt<<<2048, NTHR, 0, stream>>>(dwihh, d_wih, H3 * Hc);
  k_cvt<<<2048, NTHR, 0, stream>>>(dwhhh, d_whh, H3 * Hc);
  k_cvt<<<4096, NTHR, 0, stream>>>(swihh, s_wih, H6 * H2);
  k_cvt<<<4096, NTHR, 0, stream>>>(swhhh, s_whh, H6 * H2);
  k_cvt<<<4096, NTHR, 0, stream>>>(gwihh, g_wih, H3 * H3);
  k_cvt<<<2048, NTHR, 0, stream>>>(gwhhh, g_whh, H3 * Hc);
  k_cvt<<<1024, NTHR, 0, stream>>>(wqh, wq, Hc * Hc);
  k_cvt<<<1024, NTHR, 0, stream>>>(wkh, wk, Hc * Hc);
  k_cvt<<<1024, NTHR, 0, stream>>>(wvh, wv, Hc * Hc);
  k_gcomb<<<2048, NTHR, 0, stream>>>(gcomb, g_wih);
  k_gather<<<1024, NTHR, 0, stream>>>(berth, emb, inputs);

  // GI_d = bert @ d_wih^T + d_bih   [1024 x 3H]
  k_gemm<<<dim3(H3 / 128, TOK / 16), NTHR, 0, stream>>>(berth, Hc, dwihh, Hc, d_bih,
                                                        GId, H3, H3, Hc);
  // rnnD scan (sequential over S, cooperative grid)
  k_rnnd<<<NBLK, NTHR, 0, stream>>>(hh, dwhhh, GId, d_bhh, hf, ghw, dia, diah, bar);
  // token-parallel precomputes from dialogue
  k_gemm<<<dim3(Hc / 128, TOK / 16), NTHR, 0, stream>>>(diah, Hc, wqh, Hc, bq,
                                                        QU, Hc, Hc, Hc);
  k_gemm<<<dim3(H6 / 128, TOK / 16), NTHR, 0, stream>>>(diah, Hc, swihh + Hc, H2, s_bih,
                                                        SIU, H6, H6, Hc);
  k_gemm<<<dim3(H3 / 128, TOK / 16), NTHR, 0, stream>>>(diah, Hc, gcomb, Hc, g_bih,
                                                        GIU, H3, H3, Hc);
  // sequential attention + GRU-S + GRU-G chain (cooperative grid)
  k_mainloop<<<NBLK, NTHR, 0, stream>>>(swihh, swhhh, gwihh, gwhhh, wkh, wvh,
                                        QU, SIU, GIU, dia, Kc, Vc, scores, attnp,
                                        gis, ghs, gig, ghg, ug, dh, sp,
                                        chat_ids, spk_info, s_bhh, g_bhh, bk, bv,
                                        h0, d_h0, attn_h, out_op, out_spop, bar);
}
